// MambaBlock_89034672046299
// MI455X (gfx1250) — compile-verified, matrix operand order checked by reading, not run on hardware
//
#include <hip/hip_runtime.h>
#include <hip/hip_bf16.h>

// ---------------- Mamba block for MI455X (gfx1250, wave32, WMMA) ----------------
#define D_MODEL 1024
#define D_INNER 2048
#define D_STATE 16
#define DT_RANK 64
#define D_CONV  4
#define BATCH   2
#define SEQL    1024
#define NTOK    (BATCH * SEQL)            // 2048 rows for all GEMMs
#define XDBL_W  96                        // DT_RANK + 2*D_STATE
#define XDBL_P  128                       // padded to a multiple of 64

typedef __attribute__((ext_vector_type(16))) __bf16         bf16x16;
typedef __attribute__((ext_vector_type(16))) unsigned short u16x16;
typedef __attribute__((ext_vector_type(8)))  unsigned short u16x8;
typedef __attribute__((ext_vector_type(8)))  float          f32x8;

__device__ __forceinline__ float silu_f(float v) {
    return v / (1.0f + __expf(-v));
}

// f32 -> bf16 round-to-nearest-even (pre-pass and k_scan's bf16 output)
__device__ __forceinline__ unsigned short f2bf(float f) {
    unsigned int u = __builtin_bit_cast(unsigned int, f);
    u += 0x7FFFu + ((u >> 16) & 1u);
    return (unsigned short)(u >> 16);
}

// Async global->LDS 16-B copy (gfx1250 GLOBAL_LOAD_ASYNC_TO_LDS_B128, ASYNCcnt).
// LDS address = low 32 bits of the generic __shared__ pointer (DS-space offset).
__device__ __forceinline__ void async_ld_b128(void* lds_ptr, const void* gptr) {
    const unsigned           loff = (unsigned)(unsigned long long)lds_ptr;
    const unsigned long long gadr = (unsigned long long)gptr;
    asm volatile("global_load_async_to_lds_b128 %0, %1, off"
                 :: "v"(loff), "v"(gadr) : "memory");
}

// Async loads complete in order: waiting ASYNCcnt<=2 with 2 copies/stage
// guarantees the tile issued two stages ago has fully landed in LDS.
__device__ __forceinline__ void wait_async0() {
#if __has_builtin(__builtin_amdgcn_s_wait_asynccnt)
    __builtin_amdgcn_s_wait_asynccnt(0);
#else
    asm volatile("s_wait_asynccnt 0x0" ::: "memory");
#endif
}
__device__ __forceinline__ void wait_async_le2() {
#if __has_builtin(__builtin_amdgcn_s_wait_asynccnt)
    __builtin_amdgcn_s_wait_asynccnt(2);
#else
    asm volatile("s_wait_asynccnt 0x2" ::: "memory");
#endif
}

// LDS tile rows padded to 40 u16 (80 B): 16-B aligned b128 accesses, 20-bank
// row stride (20*r mod 64 distinct for r=0..15 -> conflict-free fragment reads).
#define BPAD 40

// ---------------------------------------------------------------------------
// GEMM: C[M,N] = act( A[M,K] @ B[K,N] + bias ), bf16 operands, f32 accumulate.
//   A  : bf16 [M][LDA]  (activations, K-contiguous rows)
//   Bt : bf16 [N][LDB]  (weights pre-transposed, K-contiguous rows)
// Compile-time dims; M = gridDim.y * 64; N % 64 == 0; K % 32 == 0, K >= 64.
// Block: 256 threads = 8 wave32s arranged 4(M) x 2(N); block tile 64x64.
// Triple-buffered LDS, 2-tile-deep async pipeline (s_wait_asynccnt <= 2).
// Per thread per K-step: 2x global_load_async_to_lds_b128, 6x ds_load_b128,
// 2x v_wmma_f32_16x16x32_bf16. ACT: 0 none, 1 softplus.
// ---------------------------------------------------------------------------
template <int LDA, int LDB, int LDC, int N, int K, int ACT>
__global__ void __launch_bounds__(256)
k_gemm(const unsigned short* __restrict__ A, const unsigned short* __restrict__ Bt,
       const float* __restrict__ bias, float* __restrict__ C) {
    static_assert(K >= 64 && K % 32 == 0 && N % 64 == 0, "tile constraints");
    __shared__ unsigned short sA[3][64][BPAD];   // [m][k] bf16
    __shared__ unsigned short sB[3][64][BPAD];   // [n][k] bf16

    const int tid   = threadIdx.x;
    const int lane  = tid & 31;
    const int wid   = tid >> 5;                  // 0..7
    const int wm    = wid & 3;                   // 0..3 (M subtile)
    const int wn    = wid >> 2;                  // 0..1 (N subtile pair)
    const int lhalf = lane >> 4;
    const int lmod  = lane & 15;
    const int off   = lhalf * 8;                 // K sub-offset for half-wave

    const int m0blk = blockIdx.y * 64;
    const int nblk  = blockIdx.x * 64;

    // staging roles: thread copies 8 K-contiguous bf16 of one A row & one Bt row
    const int sm = tid & 63;                     // A row / Bt row handled
    const int kg = (tid >> 6) * 8;               // K group: {0,8,16,24}

    const unsigned short* __restrict__ Astage = A  + (size_t)(m0blk + sm) * LDA + kg;
    const unsigned short* __restrict__ Bstage = Bt + (size_t)(nblk  + sm) * LDB + kg;

    f32x8 acc0 = {}, acc1 = {};

    auto stage = [&](int k0, int buf) {          // pure async memory -> LDS copy
        async_ld_b128(&sA[buf][sm][kg], Astage + k0);
        async_ld_b128(&sB[buf][sm][kg], Bstage + k0);
    };

    auto frag = [&](const unsigned short(*s)[BPAD], int row) -> bf16x16 {
        const u16x8 lo = *reinterpret_cast<const u16x8*>(&s[row][off]);
        const u16x8 hi = *reinterpret_cast<const u16x8*>(&s[row][off + 16]);
        const u16x16 u = __builtin_shufflevector(lo, hi,
                             0, 1, 2, 3, 4, 5, 6, 7, 8, 9, 10, 11, 12, 13, 14, 15);
        return __builtin_bit_cast(bf16x16, u);
    };

    auto compute = [&](int buf) {
        const bf16x16 af  = frag(sA[buf], wm * 16 + lmod);
        const bf16x16 bf0 = frag(sB[buf], wn * 16 + lmod);
        const bf16x16 bf1 = frag(sB[buf], wn * 16 + 32 + lmod);
        acc0 = __builtin_amdgcn_wmma_f32_16x16x32_bf16(
                   false, af, false, bf0, (short)0, acc0, false, false);
        acc1 = __builtin_amdgcn_wmma_f32_16x16x32_bf16(
                   false, af, false, bf1, (short)0, acc1, false, false);
    };

    // ---- prologue: two tiles in flight
    stage(0, 0);
    stage(32, 1);
    wait_async_le2();                // tile 0 landed (in-order completion)
    __syncthreads();

    int bufc = 0, bufs = 2;
    for (int k0 = 0; k0 < K - 64; k0 += 32) {
        stage(k0 + 64, bufs);        // tile i+2, stays in flight past barrier
        compute(bufc);               // tile i
        bufc = (bufc == 2) ? 0 : bufc + 1;
        bufs = (bufs == 2) ? 0 : bufs + 1;
        wait_async_le2();            // tile i+1 landed; tile i+2 still flying
        __syncthreads();
    }
    compute(bufc);                   // tile K/32-2
    wait_async0();                   // last tile landed
    __syncthreads();
    compute((bufc == 2) ? 0 : bufc + 1);   // tile K/32-1

    // ---- epilogue ----
    const int col0 = nblk + wn * 16 + lmod;
    const float bs0 = bias ? bias[col0] : 0.0f;
    const float bs1 = bias ? bias[col0 + 32] : 0.0f;
    #pragma unroll
    for (int v = 0; v < 8; ++v) {
        const int row = m0blk + wm * 16 + lhalf * 8 + v;   // C: VGPR v -> M=8*half+v
        float v0 = acc0[v] + bs0;
        float v1 = acc1[v] + bs1;
        if (ACT == 1) {
            v0 = (v0 > 20.0f) ? v0 : log1pf(__expf(v0));
            v1 = (v1 > 20.0f) ? v1 : log1pf(__expf(v1));
        }
        C[(size_t)row * LDC + col0]      = v0;
        C[(size_t)row * LDC + col0 + 32] = v1;
    }
}

// ---------------------------------------------------------------------------
// Pre-pass: f32 [K][NSRC] -> bf16 transposed [N][K] (zero-fill n >= NSRC).
// ---------------------------------------------------------------------------
template <int K, int N, int NSRC>
__global__ void __launch_bounds__(256)
k_cvt_t(const float* __restrict__ src, unsigned short* __restrict__ dst) {
    const int i = blockIdx.x * 256 + threadIdx.x;    // over N*K
    if (i >= N * K) return;
    const int n = i / K, k = i % K;
    const float v = (n < NSRC) ? src[(size_t)k * NSRC + n] : 0.0f;
    dst[i] = f2bf(v);
}

// Pre-pass: f32 -> bf16, same layout.
__global__ void __launch_bounds__(256)
k_cvt(const float* __restrict__ src, unsigned short* __restrict__ dst, int n) {
    const int i = blockIdx.x * 256 + threadIdx.x;
    if (i < n) dst[i] = f2bf(src[i]);
}

// ---------------------------------------------------------------------------
// Depthwise 'same' conv (kw=4, pad_left=1, pad_right=2) along L + bias + SiLU.
// ---------------------------------------------------------------------------
__global__ void __launch_bounds__(256)
k_conv_silu(const float* __restrict__ xr,
            const float* __restrict__ Kc,     // [4,1,D_INNER]
            const float* __restrict__ bc,     // [D_INNER]
            float* __restrict__ xconv) {
    const int idx = blockIdx.x * blockDim.x + threadIdx.x;   // over B*L*D_INNER
    if (idx >= NTOK * D_INNER) return;
    const int c  = idx % D_INNER;
    const int bl = idx / D_INNER;
    const int l  = bl % SEQL;
    const int b  = bl / SEQL;

    float acc = bc[c];
    #pragma unroll
    for (int w = 0; w < D_CONV; ++w) {
        const int li = l - 1 + w;            // pad_left = (4-1)/2 = 1
        if (li >= 0 && li < SEQL) {
            const float xv = xr[(size_t)(b * SEQL + li) * (2 * D_INNER) + c];
            acc += Kc[w * D_INNER + c] * xv;
        }
    }
    xconv[(size_t)bl * D_INNER + c] = silu_f(acc);
}

// ---------------------------------------------------------------------------
// Selective scan fused with output gating; emits ygated directly as bf16
// (it is consumed only by the final WMMA GEMM). f32 recurrence state.
// ---------------------------------------------------------------------------
__global__ void __launch_bounds__(256)
k_scan(const float* __restrict__ xconv,   // u      [B,L,D_INNER]
       const float* __restrict__ delta,   // Δ      [B,L,D_INNER] (post-softplus)
       const float* __restrict__ xdbl,    // [B,L,XDBL_P]: cols 64..79 = B, 80..95 = C
       const float* __restrict__ A_log,   // [D_INNER, D_STATE]
       const float* __restrict__ Dv,      // [D_INNER]
       const float* __restrict__ xr,      // res = xr[:, D_INNER:2*D_INNER]
       unsigned short* __restrict__ yg) { // bf16 [B,L,D_INNER]
    const int b = blockIdx.x >> 3;
    const int d = ((blockIdx.x & 7) << 8) + threadIdx.x;

    __shared__ float sB[D_STATE];
    __shared__ float sC[D_STATE];

    float Ad[D_STATE];
    #pragma unroll
    for (int n = 0; n < D_STATE; ++n) Ad[n] = -__expf(A_log[d * D_STATE + n]);
    const float Dd = Dv[d];

    float h[D_STATE];
    #pragma unroll
    for (int n = 0; n < D_STATE; ++n) h[n] = 0.0f;

    for (int l = 0; l < SEQL; ++l) {
        const size_t row = (size_t)(b * SEQL + l);
        __syncthreads();
        if (threadIdx.x < D_STATE) {
            sB[threadIdx.x] = xdbl[row * XDBL_P + DT_RANK + threadIdx.x];
        } else if (threadIdx.x < 2 * D_STATE) {
            sC[threadIdx.x - D_STATE] = xdbl[row * XDBL_P + DT_RANK + D_STATE +
                                             (threadIdx.x - D_STATE)];
        }
        __syncthreads();

        const float dl = delta[row * D_INNER + d];
        const float u  = xconv[row * D_INNER + d];
        const float du = dl * u;

        float acc = 0.0f;
        #pragma unroll
        for (int n = 0; n < D_STATE; ++n) {
            h[n] = __expf(dl * Ad[n]) * h[n] + du * sB[n];
            acc += h[n] * sC[n];
        }

        const float yv = acc + u * Dd;
        const float r  = xr[row * (2 * D_INNER) + D_INNER + d];
        yg[row * D_INNER + d] = f2bf(yv * silu_f(r));
    }
}

// ---------------------------------------------------------------------------
extern "C" void kernel_launch(void* const* d_in, const int* in_sizes, int n_in,
                              void* d_out, int out_size, void* d_ws, size_t ws_size,
                              hipStream_t stream) {
    const float* inputs = (const float*)d_in[0];   // [B,L,D_MODEL]
    const float* W_in   = (const float*)d_in[1];   // [D_MODEL, 2*D_INNER]
    const float* K_conv = (const float*)d_in[2];   // [4,1,D_INNER]
    const float* b_conv = (const float*)d_in[3];   // [D_INNER]
    const float* W_x    = (const float*)d_in[4];   // [D_MODEL, 96]
    const float* W_dt   = (const float*)d_in[5];   // [DT_RANK, D_INNER]
    const float* b_dt   = (const float*)d_in[6];   // [D_INNER]
    const float* A_log  = (const float*)d_in[7];   // [D_INNER, D_STATE]
    const float* Dv     = (const float*)d_in[8];   // [D_INNER]
    const float* W_out  = (const float*)d_in[9];   // [D_INNER, D_MODEL]
    const float* b_out  = (const float*)d_in[10];  // [D_MODEL]
    float* out = (float*)d_out;                    // [B,L,D_MODEL] f32

    // Workspace layout (~90 MB), fully rewritten each call.
    float* ws     = (float*)d_ws;
    float* xr     = ws;                                        // f32 [2048, 4096]
    float* xconv  = xr     + (size_t)NTOK * (2 * D_INNER);     // f32 [2048, 2048]
    float* xdbl   = xconv  + (size_t)NTOK * D_INNER;           // f32 [2048, 128]
    float* delta  = xdbl   + (size_t)NTOK * XDBL_P;            // f32 [2048, 2048]
    unsigned short* in_bf  = (unsigned short*)(delta + (size_t)NTOK * D_INNER);
    unsigned short* xd_bf  = in_bf  + (size_t)NTOK * D_MODEL;  // bf16 [2048, 128]
    unsigned short* yg_bf  = xd_bf  + (size_t)NTOK * XDBL_P;   // bf16 [2048, 2048]
    unsigned short* Wt_in  = yg_bf  + (size_t)NTOK * D_INNER;  // bf16 [4096, 1024]
    unsigned short* Wt_x   = Wt_in  + (size_t)(2 * D_INNER) * D_MODEL; // [128, 1024]
    unsigned short* Wt_dt  = Wt_x   + (size_t)XDBL_P * D_MODEL;        // [2048, 64]
    unsigned short* Wt_out = Wt_dt  + (size_t)D_INNER * DT_RANK;       // [1024, 2048]

    const dim3 blk(256);
    const int  gy = NTOK / 64;                                 // 32

    // ---- pre-pass: bf16 conversions + weight transposes (one-time per call)
    k_cvt<<<(NTOK * D_MODEL) / 256, blk, 0, stream>>>(inputs, in_bf, NTOK * D_MODEL);
    k_cvt_t<D_MODEL, 2 * D_INNER, 2 * D_INNER>
        <<<(2 * D_INNER * D_MODEL) / 256, blk, 0, stream>>>(W_in, Wt_in);
    k_cvt_t<D_MODEL, XDBL_P, XDBL_W>
        <<<(XDBL_P * D_MODEL) / 256, blk, 0, stream>>>(W_x, Wt_x);
    k_cvt_t<DT_RANK, D_INNER, D_INNER>
        <<<(D_INNER * DT_RANK) / 256, blk, 0, stream>>>(W_dt, Wt_dt);
    k_cvt_t<D_INNER, D_MODEL, D_MODEL>
        <<<(D_MODEL * D_INNER) / 256, blk, 0, stream>>>(W_out, Wt_out);

    // 1) x_and_res = inputs @ W_in                (M=2048,K=1024,N=4096)
    k_gemm<D_MODEL, D_MODEL, 2 * D_INNER, 2 * D_INNER, D_MODEL, 0>
        <<<dim3((2 * D_INNER) / 64, gy), blk, 0, stream>>>(in_bf, Wt_in, nullptr, xr);

    // 2) x_dbl = inputs @ W_x (padded)            (M=2048,K=1024,N=128)
    k_gemm<D_MODEL, D_MODEL, XDBL_P, XDBL_P, D_MODEL, 0>
        <<<dim3(XDBL_P / 64, gy), blk, 0, stream>>>(in_bf, Wt_x, nullptr, xdbl);

    // 3) x = silu(depthwise_conv(x) + b_conv)
    k_conv_silu<<<(NTOK * D_INNER) / 256, blk, 0, stream>>>(xr, K_conv, b_conv, xconv);

    // 3b) bf16 copy of x_dbl (delta-rank columns feed GEMM 4)
    k_cvt<<<(NTOK * XDBL_P) / 256, blk, 0, stream>>>(xdbl, xd_bf, NTOK * XDBL_P);

    // 4) delta = softplus(x_dbl[:, :64] @ W_dt + b_dt)   (M=2048,K=64,N=2048)
    k_gemm<XDBL_P, DT_RANK, D_INNER, D_INNER, DT_RANK, 1>
        <<<dim3(D_INNER / 64, gy), blk, 0, stream>>>(xd_bf, Wt_dt, b_dt, delta);

    // 5) selective scan + gate -> bf16
    k_scan<<<(BATCH * D_INNER) / 256, blk, 0, stream>>>(
        xconv, delta, xdbl, A_log, Dv, xr, yg_bf);

    // 6) out = ygated @ W_out + b_out             (M=2048,K=2048,N=1024)
    k_gemm<D_INNER, D_INNER, D_MODEL, D_MODEL, D_INNER, 0>
        <<<dim3(D_MODEL / 64, gy), blk, 0, stream>>>(yg_bf, Wt_out, b_out, out);
}
